// CausalSelfAttention_39298950759161
// MI455X (gfx1250) — matrix operand order repaired and checked by reading, not compiled, not run on hardware
//
#include <hip/hip_runtime.h>
#include <hip/hip_bf16.h>

// ---------------------------------------------------------------------------
// Causal sliding-window attention (B=4,S=2048,D=1024,H=16,HD=64,WINDOW=128)
// bf16 WMMA (v_wmma_f32_16x16x32_bf16) everywhere; QKV GEMM double-buffered
// through LDS; attention K-band staged with gfx1250 async-to-LDS copies.
// ---------------------------------------------------------------------------

#define BDIM 4
#define SDIM 2048
#define DDIM 1024
#define HN   16
#define HD   64
#define WIN  128

typedef __attribute__((ext_vector_type(16))) __bf16 bf16x16;
typedef __attribute__((ext_vector_type(8)))  __bf16 bf16x8;
typedef __attribute__((ext_vector_type(4)))  __bf16 bf16x4;
typedef __attribute__((ext_vector_type(8)))  float  v8f;
typedef __attribute__((ext_vector_type(4)))  float  f32x4;
typedef __attribute__((ext_vector_type(4)))  int    intx4;

union BF16Frag {
    bf16x16 v;
    bf16x8  h[2];
};

// ---- gfx1250 async global->LDS copy (guarded; falls back to sync copy) ----
#if defined(__has_builtin)
#if __has_builtin(__builtin_amdgcn_global_load_async_to_lds_b128)
#define HAVE_ASYNC_LDS 1
#endif
#endif

__device__ __forceinline__ void cp16_g2l(__bf16* lds_dst, const __bf16* g_src) {
#if defined(HAVE_ASYNC_LDS)
    __builtin_amdgcn_global_load_async_to_lds_b128(
        (__attribute__((address_space(1))) intx4*)(void*)g_src,
        (__attribute__((address_space(3))) intx4*)(void*)lds_dst,
        0, 0);
#else
    *(bf16x8*)lds_dst = *(const bf16x8*)g_src;
#endif
}

__device__ __forceinline__ void wait_async_copies() {
#if defined(HAVE_ASYNC_LDS)
#if __has_builtin(__builtin_amdgcn_s_wait_asynccnt)
    __builtin_amdgcn_s_wait_asynccnt(0);
#else
    asm volatile("s_wait_asynccnt 0" ::: "memory");
#endif
#endif
}

// ---------------------------------------------------------------------------
// Kernel 1: QKV projection.  out = hidden @ W^T + b  (M=8192, N=1024, K=1024)
// Block = 256 threads (8 waves), tile 128(M) x 128(N), K stepped by 32 with
// double-buffered LDS (global loads of step k+1 overlap WMMAs of step k).
// Wave tile 32x64 -> 8 WMMAs per K-step.  z picks {Q,K,V}; Q,K stored bf16
// (B,H,S,HD); V stored bf16 transposed (B,H,HD,S).
// ---------------------------------------------------------------------------
__global__ __launch_bounds__(256) void qkv_proj_kernel(
    const float* __restrict__ hidden,
    const float* __restrict__ Wq, const float* __restrict__ bq,
    const float* __restrict__ Wk, const float* __restrict__ bk,
    const float* __restrict__ Wv, const float* __restrict__ bv,
    __bf16* __restrict__ qo, __bf16* __restrict__ ko, __bf16* __restrict__ vto)
{
    const int z = blockIdx.z;
    const float*  W    = (z == 0) ? Wq : (z == 1) ? Wk : Wv;
    const float*  bias = (z == 0) ? bq : (z == 1) ? bk : bv;
    __bf16*       dst  = (z == 0) ? qo : (z == 1) ? ko : vto;

    // padded row stride 40 bf16 (80B) -> conflict-free striped gathers
    __shared__ __bf16 lA[2][128 * 40];
    __shared__ __bf16 lB[2][128 * 40];

    const int tid   = threadIdx.x;
    const int lane  = tid & 31;
    const int w     = tid >> 5;          // wave 0..7
    const int waveM = w >> 1;            // 0..3 (32 rows)
    const int waveN = w & 1;             // 0..1 (64 cols)
    const int m0    = blockIdx.y * 128;
    const int n0    = blockIdx.x * 128;

    const int ldRow = tid >> 1;          // 0..127 (shared by A and B loads)
    const int ldCol = (tid & 1) * 16;    // 0 / 16

    const int k0a = (lane < 16) ? 0 : 16;  // A-fragment chunk base
    const int k0b = (lane < 16) ? 0 : 16;  // B-fragment chunk base

    v8f acc[2][4] = {};
    f32x4 ra[4], rb[4];

    auto load_tiles = [&](int kc) {
        const float* ap = hidden + (size_t)(m0 + ldRow) * DDIM + kc + ldCol;
        const float* wp = W      + (size_t)(n0 + ldRow) * DDIM + kc + ldCol;
        #pragma unroll
        for (int i = 0; i < 4; ++i) ra[i] = *(const f32x4*)(ap + i * 4);
        #pragma unroll
        for (int i = 0; i < 4; ++i) rb[i] = *(const f32x4*)(wp + i * 4);
    };
    auto store_tiles = [&](int buf) {
        #pragma unroll
        for (int i = 0; i < 4; ++i)
            *(bf16x4*)&lA[buf][ldRow * 40 + ldCol + i * 4] = __builtin_convertvector(ra[i], bf16x4);
        #pragma unroll
        for (int i = 0; i < 4; ++i)
            *(bf16x4*)&lB[buf][ldRow * 40 + ldCol + i * 4] = __builtin_convertvector(rb[i], bf16x4);
    };

    load_tiles(0);
    store_tiles(0);
    __syncthreads();

    int cur = 0;
    for (int kc = 0; kc < DDIM; kc += 32) {
        const bool more = (kc + 32) < DDIM;
        if (more) load_tiles(kc + 32);           // overlaps WMMAs below

        BF16Frag a[2], b[4];
        #pragma unroll
        for (int sm = 0; sm < 2; ++sm) {
            const __bf16* p = &lA[cur][(waveM * 32 + sm * 16 + (lane & 15)) * 40];
            a[sm].h[0] = *(const bf16x8*)(p + k0a);
            a[sm].h[1] = *(const bf16x8*)(p + k0a + 8);
        }
        #pragma unroll
        for (int sn = 0; sn < 4; ++sn) {
            const __bf16* p = &lB[cur][(waveN * 64 + sn * 16 + (lane & 15)) * 40];
            b[sn].h[0] = *(const bf16x8*)(p + k0b);
            b[sn].h[1] = *(const bf16x8*)(p + k0b + 8);
        }
        #pragma unroll
        for (int sm = 0; sm < 2; ++sm)
            #pragma unroll
            for (int sn = 0; sn < 4; ++sn)
                acc[sm][sn] = __builtin_amdgcn_wmma_f32_16x16x32_bf16(
                    false, a[sm].v, false, b[sn].v, (short)0, acc[sm][sn], false, false);

        if (more) {
            store_tiles(cur ^ 1);                // other buffer: no conflict
            __syncthreads();
            cur ^= 1;
        }
    }

    // epilogue: bias add + scatter to bf16 Q/K (row-major) or V^T
    #pragma unroll
    for (int sm = 0; sm < 2; ++sm) {
        #pragma unroll
        for (int sn = 0; sn < 4; ++sn) {
            const int n     = n0 + waveN * 64 + sn * 16 + (lane & 15);
            const float bv_ = bias[n];
            const int mbase = m0 + waveM * 32 + sm * 16 + ((lane < 16) ? 0 : 8);
            const int bi    = mbase >> 11;           // / SDIM
            const int sbase = mbase & (SDIM - 1);
            const int h     = n >> 6;
            const int hd    = n & 63;
            if (z < 2) {
                #pragma unroll
                for (int r = 0; r < 8; ++r) {
                    float val = acc[sm][sn][r] + bv_;
                    size_t off = (((size_t)bi * HN + h) * SDIM + (sbase + r)) * HD + hd;
                    dst[off] = (__bf16)val;
                }
            } else {
                bf16x8 pack;
                #pragma unroll
                for (int r = 0; r < 8; ++r)
                    pack[r] = (__bf16)(acc[sm][sn][r] + bv_);
                size_t off = (((size_t)bi * HN + h) * HD + hd) * SDIM + sbase;
                *(bf16x8*)(dst + off) = pack;        // 16B contiguous (8 tokens)
            }
        }
    }
}

// ---------------------------------------------------------------------------
// Kernel 2: banded attention.  Block = 4 waves = queries [T0, T0+64); the
// shared 208-key K band [T0-128, T0+80) is staged once into LDS via the
// gfx1250 async-to-LDS path.  Each wave owns one 16-query tile with a fixed
// 160-key (10-tile) band -> exact softmax, no online rescale.
// ---------------------------------------------------------------------------
#define KBAND 208                      // 13 key tiles of 16
#define KSTR  72                       // padded LDS row stride (bf16)

__global__ __launch_bounds__(128) void attn_window_kernel(
    const __bf16* __restrict__ q, const __bf16* __restrict__ k,
    const __bf16* __restrict__ vt, const float* __restrict__ amask,
    float* __restrict__ out)
{
    __shared__ __bf16 lK[KBAND * KSTR];      // 29.3 KB: K band, key-major
    __shared__ __bf16 lP[4][16 * 168];       // 21.0 KB: per-wave P staging

    const int tid  = threadIdx.x;
    const int lane = tid & 31;
    const int w    = tid >> 5;
    const int t64  = blockIdx.x & 31;              // S/64 = 32 query blocks
    const int h    = (blockIdx.x >> 5) & (HN - 1);
    const int b    = blockIdx.x >> 9;
    const int T0   = t64 * 64;
    const int t0   = T0 + w * 16;                  // this wave's query tile
    const int kb0  = T0 - WIN;                     // band base (mult of 16)
    const float scale = 0.125f;                    // 1/sqrt(64)

    const __bf16* qb = q  + ((size_t)b * HN + h) * SDIM * HD;
    const __bf16* kb = k  + ((size_t)b * HN + h) * SDIM * HD;
    const __bf16* vb = vt + ((size_t)b * HN + h) * HD * SDIM;
    const float*  mb = amask + (size_t)b * SDIM;

    // ---- stage K band into LDS: 208 rows x 64 hd, 8x16B per row ----------
    #pragma unroll
    for (int p = 0; p < 13; ++p) {
        const int t   = p * 128 + tid;             // 0..1663
        const int row = t >> 3;                    // 0..207
        const int seg = (t & 7) * 8;               // hd chunk
        int key = kb0 + row;
        key = (key < 0) ? 0 : (key >= SDIM ? SDIM - 1 : key);   // OOB rows masked later
        cp16_g2l(&lK[row * KSTR + seg], kb + (size_t)key * HD + seg);
    }

    // ---- Q A-fragments (overlap with async K staging) ---------------------
    BF16Frag aq[2];
    {
        const __bf16* p  = qb + (size_t)(t0 + (lane & 15)) * HD;
        const int     k0 = (lane < 16) ? 0 : 16;
        aq[0].h[0] = *(const bf16x8*)(p + k0);
        aq[0].h[1] = *(const bf16x8*)(p + k0 + 8);
        aq[1].h[0] = *(const bf16x8*)(p + 32 + k0);
        aq[1].h[1] = *(const bf16x8*)(p + 32 + k0 + 8);
    }

    wait_async_copies();
    __syncthreads();

    const int relbase = t0 - kb0;                  // 128 + 16*w: query tile offset in band
    float sc[10][8];

    // ---- scores: Q @ K^T over this wave's 160-key sub-band ---------------
    #pragma unroll
    for (int j = 0; j < 10; ++j) {
        v8f acc = {};
        const int rel = relbase + j * 16 + (lane & 15);   // row in lK
        const __bf16* kp = &lK[rel * KSTR];
        const int k0 = (lane < 16) ? 0 : 16;
        #pragma unroll
        for (int ka = 0; ka < 2; ++ka) {
            BF16Frag bk_;
            bk_.h[0] = *(const bf16x8*)(kp + ka * 32 + k0);
            bk_.h[1] = *(const bf16x8*)(kp + ka * 32 + k0 + 8);
            acc = __builtin_amdgcn_wmma_f32_16x16x32_bf16(
                false, aq[ka].v, false, bk_.v, (short)0, acc, false, false);
        }
        const int jk = t0 - WIN + j * 16 + (lane & 15);   // absolute key index
        const float maskadd = (jk >= 0 && jk < SDIM) ? mb[jk] : 0.0f;
        #pragma unroll
        for (int r = 0; r < 8; ++r) {
            const int i = t0 + r + ((lane < 16) ? 0 : 8);
            const bool ok = (jk >= 0) && (jk < SDIM) && (jk <= i) && (i - jk <= WIN);
            sc[j][r] = ok ? (acc[r] * scale + maskadd) : -1e30f;
        }
    }

    // ---- exact softmax: row reductions across 16-lane halves (wave32) ----
    #pragma unroll
    for (int r = 0; r < 8; ++r) {
        float m = -1e30f;
        #pragma unroll
        for (int j = 0; j < 10; ++j) m = fmaxf(m, sc[j][r]);
        #pragma unroll
        for (int x = 1; x < 16; x <<= 1) m = fmaxf(m, __shfl_xor(m, x, 32));
        float s = 0.0f;
        #pragma unroll
        for (int j = 0; j < 10; ++j) { sc[j][r] = __expf(sc[j][r] - m); s += sc[j][r]; }
        #pragma unroll
        for (int x = 1; x < 16; x <<= 1) s += __shfl_xor(s, x, 32);
        const float inv = 1.0f / s;
        #pragma unroll
        for (int j = 0; j < 10; ++j) sc[j][r] *= inv;
    }

    // ---- stage P (C-layout -> A-layout) through LDS as bf16 ---------------
    __bf16* pl = &lP[w][0];
    {
        const int dk = lane & 15;
        const int rb = (lane < 16) ? 0 : 8;
        #pragma unroll
        for (int j = 0; j < 10; ++j)
            #pragma unroll
            for (int r = 0; r < 8; ++r)
                pl[(r + rb) * 168 + j * 16 + dk] = (__bf16)sc[j][r];
    }
    __syncthreads();

    // ---- ctx = P @ V (K=160 in 5 steps; V from (B,H,HD,S), contiguous) ----
    v8f ctx[4] = {};
    #pragma unroll
    for (int kc = 0; kc < 5; ++kc) {
        BF16Frag ap;
        const int k0 = (lane < 16) ? 0 : 16;
        const __bf16* pp = pl + (lane & 15) * 168 + kc * 32;
        ap.h[0] = *(const bf16x8*)(pp + k0);
        ap.h[1] = *(const bf16x8*)(pp + k0 + 8);

        int keyoff = t0 - WIN + kc * 32 + ((lane < 16) ? 0 : 16);
        keyoff = (keyoff < 0) ? 0 : (keyoff > SDIM - 16 ? SDIM - 16 : keyoff);
        #pragma unroll
        for (int nn = 0; nn < 4; ++nn) {
            const int hd = nn * 16 + (lane & 15);
            const __bf16* vp = vb + (size_t)hd * SDIM + keyoff;
            BF16Frag bv_;
            bv_.h[0] = *(const bf16x8*)(vp);
            bv_.h[1] = *(const bf16x8*)(vp + 8);
            ctx[nn] = __builtin_amdgcn_wmma_f32_16x16x32_bf16(
                false, ap.v, false, bv_.v, (short)0, ctx[nn], false, false);
        }
    }

    // ---- store: out[b][t][h*64+hd] fp32 -----------------------------------
    #pragma unroll
    for (int nn = 0; nn < 4; ++nn) {
        const int hd = nn * 16 + (lane & 15);
        const int rb = (lane < 16) ? 0 : 8;
        #pragma unroll
        for (int r = 0; r < 8; ++r) {
            const int t = t0 + r + rb;
            out[((size_t)b * SDIM + t) * DDIM + h * HD + hd] = ctx[nn][r];
        }
    }
}

// ---------------------------------------------------------------------------
extern "C" void kernel_launch(void* const* d_in, const int* in_sizes, int n_in,
                              void* d_out, int out_size, void* d_ws, size_t ws_size,
                              hipStream_t stream) {
    const float* hidden = (const float*)d_in[0];
    const float* amask  = (const float*)d_in[1];
    const float* Wq     = (const float*)d_in[2];
    const float* bq     = (const float*)d_in[3];
    const float* Wk     = (const float*)d_in[4];
    const float* bk     = (const float*)d_in[5];
    const float* Wv     = (const float*)d_in[6];
    const float* bv     = (const float*)d_in[7];
    float* out          = (float*)d_out;

    const size_t elems = (size_t)BDIM * SDIM * DDIM;   // 8M per tensor
    __bf16* qw = (__bf16*)d_ws;
    __bf16* kw = qw + elems;
    __bf16* vw = kw + elems;                            // 48 MB total scratch

    // QKV projections: grid = (N-tiles=8, M-tiles=64, 3 matrices)
    qkv_proj_kernel<<<dim3(8, 64, 3), 256, 0, stream>>>(
        hidden, Wq, bq, Wk, bk, Wv, bv, qw, kw, vw);

    // attention: B*H*(S/64) = 2048 blocks, 4 waves (4 query tiles) each
    attn_window_kernel<<<dim3(2048), 128, 0, stream>>>(qw, kw, vw, amask, out);
}
